// SingleHeadAttention_2516850835846
// MI455X (gfx1250) — matrix operand order repaired and checked by reading, not compiled, not run on hardware
//
#include <hip/hip_runtime.h>
#include <hip/hip_bf16.h>

// ---------------------------------------------------------------------------
// Single-head causal attention, head_size = n_embd = 1024, B=8, T=2048.
// All GEMMs via v_wmma_f32_16x16x32_f16 (fp32 accumulate).
//   phase 0: vectorized f32 -> f16 conversion of x and Wq/Wk/Wv
//   phase 1: projections q = x Wq^T, k = x Wk^T (row-major), v^T (transposed)
//            - x row-tile staged in LDS via global_load_async_to_lds_b128
//            - W B-fragments ping-pong double-buffered
//   phase 2: flash attention (Br=16 queries/block, Bc=64 keys/iter),
//            8 waves each own a 128-wide slice of the head dimension,
//            fused cross-wave reduce + softmax over all 256 threads,
//            32 WMMAs per wave between each barrier pair.
// Workspace layout (needs ~134 MB).
// ---------------------------------------------------------------------------

typedef _Float16 v16h __attribute__((ext_vector_type(16)));
typedef _Float16 v8h  __attribute__((ext_vector_type(8)));
typedef float    v8f  __attribute__((ext_vector_type(8)));
typedef float    v4f  __attribute__((ext_vector_type(4)));

__device__ __forceinline__ v8f zero8() {
  v8f z;
#pragma unroll
  for (int i = 0; i < 8; ++i) z[i] = 0.0f;
  return z;
}

__device__ __forceinline__ v8f wmma_f16(v16h a, v16h b, v8f c) {
  return __builtin_amdgcn_wmma_f32_16x16x32_f16(false, a, false, b, (short)0, c,
                                                false, false);
}

// CDNA5 async copy: global -> LDS, 16 bytes per lane, tracked by ASYNCcnt.
__device__ __forceinline__ void async_copy_b128(unsigned int lds_byte_off,
                                                const void* gptr) {
  asm volatile("global_load_async_to_lds_b128 %0, %1, off"
               :
               : "v"(lds_byte_off), "v"(gptr)
               : "memory");
}

__device__ __forceinline__ void wait_async0() {
  asm volatile("s_wait_asynccnt 0x0" ::: "memory");
}

// A fragment: 16x32 (MxK) f16. Lane L<16: row M=L, K={0..7,16..23};
// lane L>=16: row M=L-16, K={8..15,24..31}. Two 16B loads per lane.
__device__ __forceinline__ v16h load_frag_a(const _Float16* base, int ld, int lane) {
  const int m  = lane & 15;
  const int kb = (lane & 16) ? 8 : 0;
  const _Float16* p = base + (size_t)m * ld + kb;
  v8h lo = *reinterpret_cast<const v8h*>(p);
  v8h hi = *reinterpret_cast<const v8h*>(p + 16);
  v16h r;
#pragma unroll
  for (int i = 0; i < 8; ++i) { r[i] = lo[i]; r[i + 8] = hi[i]; }
  return r;
}

// B fragment: 32x16 (KxN) f16. Lane n<16: column N=n, K={0..7,16..23}; lane>=16:
// column N=n-16, K={8..15,24..31}. base indexes src[n][k]; two 16B loads per lane.
__device__ __forceinline__ v16h load_frag_b(const _Float16* base, int ld, int lane) {
  const int n  = lane & 15;
  const int kb = (lane & 16) ? 8 : 0;
  const _Float16* p = base + (size_t)n * ld + kb;
  v8h lo = *reinterpret_cast<const v8h*>(p);
  v8h hi = *reinterpret_cast<const v8h*>(p + 16);
  v16h r;
#pragma unroll
  for (int i = 0; i < 8; ++i) { r[i] = lo[i]; r[i + 8] = hi[i]; }
  return r;
}

// ---------------------------------------------------------------------------
// Vectorized f32 -> f16 (8 elements per thread-iteration). n8 = n / 8.
__global__ void cvt_f32_f16_kernel(const float* __restrict__ in,
                                   _Float16* __restrict__ out, int n8) {
  int i = blockIdx.x * blockDim.x + threadIdx.x;
  const int stride = gridDim.x * blockDim.x;
  for (; i < n8; i += stride) {
    const v4f* p = reinterpret_cast<const v4f*>(in + (size_t)i * 8);
    v4f a = p[0], b = p[1];
    v8h o;
#pragma unroll
    for (int c = 0; c < 4; ++c) { o[c] = (_Float16)a[c]; o[c + 4] = (_Float16)b[c]; }
    *reinterpret_cast<v8h*>(out + (size_t)i * 8) = o;
  }
}

// ---------------------------------------------------------------------------
// y[row][h] = sum_c xh[row][c] * Wh[h][c]  (rows = 16384, h,c in [0,1024))
// Block: 256 threads (8 waves); tile 16 rows x 512 cols; wave tile 16x64.
__global__ __launch_bounds__(256) void proj_kernel(
    const _Float16* __restrict__ xh, const _Float16* __restrict__ Wh,
    _Float16* __restrict__ out, int transpose) {
  __shared__ __align__(16) _Float16 xs[16][1024];  // 32 KB x row-tile

  const int lane = threadIdx.x & 31;
  const int w    = threadIdx.x >> 5;
  const int m0   = blockIdx.x * 16;
  const int c0   = blockIdx.y * 512 + w * 64;

  // ---- async-stage the x tile (16 x 1024 f16 = 32 KB) into LDS ----
  {
    const _Float16* src = xh + (size_t)m0 * 1024;
    const unsigned int lds_base = (unsigned int)(uintptr_t)(&xs[0][0]);
#pragma unroll
    for (int i = 0; i < 8; ++i) {
      const int elem = (threadIdx.x + i * 256) * 8;  // 8 f16 = 16 B per chunk
      async_copy_b128(lds_base + elem * 2, src + elem);
    }
    wait_async0();
  }
  __syncthreads();

  v8f acc[4];
#pragma unroll
  for (int t = 0; t < 4; ++t) acc[t] = zero8();

  const _Float16* xls = &xs[0][0];
  const _Float16* wb[4];
#pragma unroll
  for (int t = 0; t < 4; ++t) wb[t] = Wh + (size_t)(c0 + t * 16) * 1024;

  v16h b0[4], b1[4];
#pragma unroll
  for (int t = 0; t < 4; ++t) b0[t] = load_frag_b(wb[t], 1024, lane);

  for (int k0 = 0; k0 < 1024; k0 += 64) {
#pragma unroll
    for (int t = 0; t < 4; ++t) b1[t] = load_frag_b(wb[t] + k0 + 32, 1024, lane);
    {
      v16h a = load_frag_a(xls + k0, 1024, lane);
#pragma unroll
      for (int t = 0; t < 4; ++t) acc[t] = wmma_f16(a, b0[t], acc[t]);
    }
    if (k0 + 64 < 1024) {
#pragma unroll
      for (int t = 0; t < 4; ++t) b0[t] = load_frag_b(wb[t] + k0 + 64, 1024, lane);
    }
    {
      v16h a = load_frag_a(xls + k0 + 32, 1024, lane);
#pragma unroll
      for (int t = 0; t < 4; ++t) acc[t] = wmma_f16(a, b1[t], acc[t]);
    }
  }

  const int Mo = (lane & 16) ? 8 : 0;
  const int N  = lane & 15;
#pragma unroll
  for (int t = 0; t < 4; ++t) {
    const int h = c0 + t * 16 + N;
#pragma unroll
    for (int r = 0; r < 8; ++r) {
      const int grow = m0 + Mo + r;
      if (!transpose) {
        out[(size_t)grow * 1024 + h] = (_Float16)acc[t][r];
      } else {
        const int bb = grow >> 11;       // batch
        const int tt = grow & 2047;      // position
        out[(size_t)bb * (1024 * 2048) + (size_t)h * 2048 + tt] =
            (_Float16)acc[t][r];
      }
    }
  }
}

// ---------------------------------------------------------------------------
// Flash attention. Grid (T/16, B), block 256 (8 waves). Bc = 64 keys/iter.
// Wave w owns head-dim slice [w*128, w*128+128).
__global__ __launch_bounds__(256) void attn_kernel(
    const _Float16* __restrict__ qh, const _Float16* __restrict__ kh,
    const _Float16* __restrict__ vth, float* __restrict__ out) {
  __shared__ float    Sp[8][16][64];                 // per-wave partial S (32 KB)
  __shared__ __align__(16) _Float16 Pb[16][64];      // exp(S - m_new) in f16 (2 KB)
  __shared__ float    alphaS[16];
  __shared__ float    mS[16];
  __shared__ float    lS[16];

  const int lane = threadIdx.x & 31;
  const int w    = threadIdx.x >> 5;
  const int m0   = blockIdx.x * 16;
  const int b    = blockIdx.y;

  const _Float16* q  = qh  + (size_t)b * 2048 * 1024;
  const _Float16* k  = kh  + (size_t)b * 2048 * 1024;
  const _Float16* vt = vth + (size_t)b * 1024 * 2048;

  if (w == 0 && lane < 16) { mS[lane] = -__builtin_inff(); lS[lane] = 0.0f; }

  // cache this block's q slice (16 x 128) as 4 A-fragments (32 VGPRs)
  v16h qa[4];
#pragma unroll
  for (int kk = 0; kk < 4; ++kk)
    qa[kk] = load_frag_a(q + (size_t)m0 * 1024 + w * 128 + kk * 32, 1024, lane);

  v8f o[8];
#pragma unroll
  for (int t = 0; t < 8; ++t) o[t] = zero8();

  const int   Mo    = (lane & 16) ? 8 : 0;
  const int   N     = lane & 15;
  const float scale = 0.03125f;  // 1/sqrt(1024)
  const int   jmax  = (m0 + 15) >> 6;  // 64-key blocks

  // prime the K fragment stream (j = 0, kk = 0): 4 key sub-tiles of 16
  v16h kb[4];
#pragma unroll
  for (int t = 0; t < 4; ++t)
    kb[t] = load_frag_b(k + (size_t)(t * 16) * 1024 + w * 128, 1024, lane);

  __syncthreads();

  for (int j = 0; j <= jmax; ++j) {
    const _Float16* kbase = k + (size_t)(j * 64) * 1024 + w * 128;

    // ---- partial S over this wave's 128-dim slice: 16x64 (16 WMMAs) ----
    v8f s[4];
#pragma unroll
    for (int t = 0; t < 4; ++t) s[t] = zero8();
#pragma unroll
    for (int kk = 0; kk < 4; ++kk) {
      v16h nb[4];
      if (kk < 3) {
#pragma unroll
        for (int t = 0; t < 4; ++t)
          nb[t] = load_frag_b(kbase + (size_t)(t * 16) * 1024 + (kk + 1) * 32,
                              1024, lane);
      }
#pragma unroll
      for (int t = 0; t < 4; ++t) s[t] = wmma_f16(qa[kk], kb[t], s[t]);
      if (kk < 3) {
#pragma unroll
        for (int t = 0; t < 4; ++t) kb[t] = nb[t];
      }
    }
#pragma unroll
    for (int t = 0; t < 4; ++t) {
#pragma unroll
      for (int r = 0; r < 8; ++r) Sp[w][Mo + r][t * 16 + N] = s[t][r];
    }
    // prefetch next key-block's first K fragments (hidden under barriers)
    if (j < jmax) {
      const _Float16* kn = k + (size_t)((j + 1) * 64) * 1024 + w * 128;
#pragma unroll
      for (int t = 0; t < 4; ++t)
        kb[t] = load_frag_b(kn + (size_t)(t * 16) * 1024, 1024, lane);
    }
    __syncthreads();

    // ---- fused cross-wave reduce + online softmax, all 256 threads ----
    // wave w handles rows 2w (lanes 0-15) and 2w+1 (lanes 16-31);
    // each lane owns a 4-column group; row max/sum via 16-lane shfl_xor.
    {
      const int row  = 2 * w + (lane >> 4);
      const int cp   = (lane & 15) * 4;
      const int qrow = m0 + row;
      v4f acc4;
#pragma unroll
      for (int c = 0; c < 4; ++c) acc4[c] = 0.0f;
#pragma unroll
      for (int ww = 0; ww < 8; ++ww) {
        v4f part = *reinterpret_cast<const v4f*>(&Sp[ww][row][cp]);
#pragma unroll
        for (int c = 0; c < 4; ++c) acc4[c] += part[c];
      }
      const int jc = j * 64;
      float a[4];
#pragma unroll
      for (int c = 0; c < 4; ++c) {
        a[c] = acc4[c] * scale;
        if (jc + cp + c > qrow) a[c] = -__builtin_inff();
      }
      float rm = fmaxf(fmaxf(a[0], a[1]), fmaxf(a[2], a[3]));
#pragma unroll
      for (int off = 1; off <= 8; off <<= 1)
        rm = fmaxf(rm, __shfl_xor(rm, off, 32));
      const float mo = mS[row];
      const float mn = fmaxf(mo, rm);
      const float al = __expf(mo - mn);  // 0 on first visit (mo = -inf)
      float p[4], ps = 0.0f;
#pragma unroll
      for (int c = 0; c < 4; ++c) { p[c] = __expf(a[c] - mn); ps += p[c]; }
#pragma unroll
      for (int off = 1; off <= 8; off <<= 1)
        ps += __shfl_xor(ps, off, 32);
      union { _Float16 h[4]; uint2 u; } pk;
#pragma unroll
      for (int c = 0; c < 4; ++c) pk.h[c] = (_Float16)p[c];
      reinterpret_cast<uint2*>(&Pb[row][0])[lane & 15] = pk.u;
      if ((lane & 15) == 0) {
        mS[row]     = mn;
        lS[row]     = lS[row] * al + ps;
        alphaS[row] = al;
      }
    }
    // prefetch first V fragment pair for this key-block (hidden under barrier)
    const _Float16* vbase = vt + (size_t)(w * 128) * 2048 + j * 64;
    v16h vb0 = load_frag_b(vbase, 2048, lane);
    v16h vb1 = load_frag_b(vbase + 32, 2048, lane);
    __syncthreads();

    // ---- rescale O and accumulate P @ V (two WMMAs per O tile, K=64) ----
    float av[8];
#pragma unroll
    for (int r = 0; r < 8; ++r) av[r] = alphaS[Mo + r];
    v16h pa0 = load_frag_a((const _Float16*)&Pb[0][0], 64, lane);
    v16h pa1 = load_frag_a((const _Float16*)&Pb[0][0] + 32, 64, lane);
#pragma unroll
    for (int t = 0; t < 8; ++t) {
      v16h nv0, nv1;
      if (t < 7) {
        nv0 = load_frag_b(vbase + (size_t)((t + 1) * 16) * 2048, 2048, lane);
        nv1 = load_frag_b(vbase + (size_t)((t + 1) * 16) * 2048 + 32, 2048, lane);
      }
#pragma unroll
      for (int r = 0; r < 8; ++r) o[t][r] *= av[r];
      o[t] = wmma_f16(pa0, vb0, o[t]);
      o[t] = wmma_f16(pa1, vb1, o[t]);
      if (t < 7) { vb0 = nv0; vb1 = nv1; }
    }
    // next iteration's first barrier orders these reads vs. future writes
  }

  // ---- epilogue: normalize by l and store fp32 ----
  float linv[8];
#pragma unroll
  for (int r = 0; r < 8; ++r) linv[r] = 1.0f / lS[Mo + r];
  float* ob = out + (size_t)b * 2048 * 1024 + (size_t)m0 * 1024 + w * 128;
#pragma unroll
  for (int t = 0; t < 8; ++t) {
#pragma unroll
    for (int r = 0; r < 8; ++r) {
      ob[(size_t)(Mo + r) * 1024 + t * 16 + N] = o[t][r] * linv[r];
    }
  }
}

// ---------------------------------------------------------------------------
extern "C" void kernel_launch(void* const* d_in, const int* in_sizes, int n_in,
                              void* d_out, int out_size, void* d_ws,
                              size_t ws_size, hipStream_t stream) {
  (void)in_sizes; (void)n_in; (void)out_size; (void)ws_size;

  const float* x  = (const float*)d_in[0];
  const float* Wk = (const float*)d_in[1];
  const float* Wq = (const float*)d_in[2];
  const float* Wv = (const float*)d_in[3];
  float* out = (float*)d_out;

  char* ws = (char*)d_ws;
  _Float16* xh  = (_Float16*)(ws + ((size_t)0 << 20));
  _Float16* qh  = (_Float16*)(ws + ((size_t)32 << 20));
  _Float16* kh  = (_Float16*)(ws + ((size_t)64 << 20));
  _Float16* vth = (_Float16*)(ws + ((size_t)96 << 20));
  _Float16* wqh = (_Float16*)(ws + ((size_t)128 << 20));
  _Float16* wkh = (_Float16*)(ws + ((size_t)130 << 20));
  _Float16* wvh = (_Float16*)(ws + ((size_t)132 << 20));

  const int nx8 = (8 * 2048 * 1024) / 8;
  const int nw8 = (1024 * 1024) / 8;
  cvt_f32_f16_kernel<<<2048, 256, 0, stream>>>(x, xh, nx8);
  cvt_f32_f16_kernel<<<512, 256, 0, stream>>>(Wq, wqh, nw8);
  cvt_f32_f16_kernel<<<512, 256, 0, stream>>>(Wk, wkh, nw8);
  cvt_f32_f16_kernel<<<512, 256, 0, stream>>>(Wv, wvh, nw8);

  dim3 pgrid(1024, 2);  // 16384/16 row tiles x 2 column halves
  proj_kernel<<<pgrid, 256, 0, stream>>>(xh, wqh, qh, 0);
  proj_kernel<<<pgrid, 256, 0, stream>>>(xh, wkh, kh, 0);
  proj_kernel<<<pgrid, 256, 0, stream>>>(xh, wvh, vth, 1);

  attn_kernel<<<dim3(128, 8), 256, 0, stream>>>(qh, kh, vth, out);
}
